// CountBasedReward_44023414784108
// MI455X (gfx1250) — matrix operand order repaired and checked by reading, not compiled, not run on hardware
//
#include <hip/hip_runtime.h>
#include <hip/hip_bf16.h>
#include <math.h>

// ---- problem constants (from reference) ----
#define B_ROWS   2048          // batch
#define D_DIM    28224         // 4*84*84
#define K_BITS   256           // SimHash bits
#define BLK_ROWS 128           // rows per workgroup in GEMM
#define KT       32            // K per wmma step (bf16)
#define KSLICES  18            // split-K slices: 18 * 49 * 32 = 28224
#define KSTEPS   49            // wmma K-steps per slice
#define SF_STRIDE 40           // padded LDS stride (bf16 elems): 80B, conflict-free, 16B aligned
#define SA_STRIDE 40

typedef __attribute__((ext_vector_type(16))) __bf16       v16bf;
typedef __attribute__((ext_vector_type(8)))  float        v8f;
typedef __attribute__((ext_vector_type(4)))  unsigned int u32x4;
typedef __attribute__((ext_vector_type(4)))  int          i32x4;
typedef __attribute__((ext_vector_type(4)))  float        f32x4;
typedef __attribute__((ext_vector_type(4)))  __bf16       bf16x4;

union Frag { v16bf v; u32x4 u[2]; };

// ---- gfx1250 async global->LDS (signature learned from round-2 diagnostics) ----
#if __has_builtin(__builtin_amdgcn_global_load_async_to_lds_b128)
#define HAVE_ASYNC_LDS 1
#endif

__device__ __forceinline__ void async_copy_b128(const unsigned int* gsrc,
                                                unsigned int* lds_dst) {
#ifdef HAVE_ASYNC_LDS
  __builtin_amdgcn_global_load_async_to_lds_b128(
      (__attribute__((address_space(1))) i32x4*)gsrc,
      (__attribute__((address_space(3))) i32x4*)lds_dst,
      /*imm offset=*/0, /*cpol=*/0);
#else
  *(u32x4*)lds_dst = *(const u32x4*)gsrc;
#endif
}

__device__ __forceinline__ void async_wait0() {
#ifdef HAVE_ASYNC_LDS
#if __has_builtin(__builtin_amdgcn_s_wait_asynccnt)
  __builtin_amdgcn_s_wait_asynccnt(0);
#else
  asm volatile("s_wait_asynccnt 0" ::: "memory");
#endif
#endif
}

// ============================================================================
// Kernel 1: split-K partial GEMM.
//   P[slice][b][j] = sum over K-slice of feats[b,:] . A[j,:]   (bf16 WMMA)
// grid = (16 row-blocks, 18 K-slices) x 256 threads (8 waves).
// Block tile: 128 rows x 256 cols x 1568 K. Wave tile: 32 rows x 128 cols
// (2 A-frags + 8 B-frags -> 16 WMMA per K-step, each B-frag used twice).
// ============================================================================
__global__ __launch_bounds__(256)
void simhash_gemm_bf16_wmma(const float* __restrict__ feats,
                            const float* __restrict__ Amat,
                            float* __restrict__ P) {
  __shared__ __bf16 sF[BLK_ROWS * SF_STRIDE];   // 10240 B
  __shared__ __bf16 sA[K_BITS  * SA_STRIDE];    // 20480 B

  const int tid    = threadIdx.x;
  const int wave   = tid >> 5;        // 0..7
  const int lane   = tid & 31;
  const int laneLo = lane & 15;
  const int hiSel  = lane >> 4;       // 0 or 1
  const int waveR  = wave >> 1;       // 0..3 : 32-row group
  const int waveC  = wave & 1;        // 0..1 : 128-col group
  const int rowBlock = blockIdx.x * BLK_ROWS;
  const int kBase    = blockIdx.y * (KSTEPS * KT);

  // cooperative staging pattern: 32 rows per pass, 8 segs of 4 floats per row
  const int ldr = tid >> 3;           // 0..31
  const int seg = tid & 7;            // 0..7

  v8f acc[2][8] = {};

  #pragma unroll 1
  for (int ks = 0; ks < KSTEPS; ++ks) {
    const int k0 = kBase + ks * KT;
    __syncthreads();                       // previous tile consumed by all waves
    // ---- stage feats tile [128 x 32], f32 -> bf16 on the fly ----
    #pragma unroll
    for (int p = 0; p < 4; ++p) {
      f32x4 f = *(const f32x4*)(feats + (size_t)(rowBlock + p * 32 + ldr) * D_DIM + k0 + seg * 4);
      bf16x4 h;
      h.x = (__bf16)f.x; h.y = (__bf16)f.y; h.z = (__bf16)f.z; h.w = (__bf16)f.w;
      *(bf16x4*)&sF[(p * 32 + ldr) * SF_STRIDE + seg * 4] = h;
    }
    // ---- stage A tile [256 x 32] ----
    #pragma unroll
    for (int p = 0; p < 8; ++p) {
      f32x4 f = *(const f32x4*)(Amat + (size_t)(p * 32 + ldr) * D_DIM + k0 + seg * 4);
      bf16x4 h;
      h.x = (__bf16)f.x; h.y = (__bf16)f.y; h.z = (__bf16)f.z; h.w = (__bf16)f.w;
      *(bf16x4*)&sA[(p * 32 + ldr) * SA_STRIDE + seg * 4] = h;
    }
    // ---- L2/L0 prefetch of the next K-tile (global_prefetch_b8) ----
    if (ks + 1 < KSTEPS) {
      #pragma unroll
      for (int p = 0; p < 4; ++p)
        __builtin_prefetch(feats + (size_t)(rowBlock + p * 32 + ldr) * D_DIM + k0 + KT + seg * 4, 0, 3);
      #pragma unroll
      for (int p = 0; p < 8; ++p)
        __builtin_prefetch(Amat + (size_t)(p * 32 + ldr) * D_DIM + k0 + KT + seg * 4, 0, 3);
    }
    __syncthreads();                       // staging complete

    // A-fragments (ISA 16-bit A 16x32 layout):
    // lanes 0-15: K 0..7 then 16..23 ; lanes 16-31: K 8..15 then 24..31
    Frag a0, a1;
    const int ar0 = waveR * 32 + laneLo;
    a0.u[0] = *(const u32x4*)&sF[ar0 * SF_STRIDE + hiSel * 8];
    a0.u[1] = *(const u32x4*)&sF[ar0 * SF_STRIDE + 16 + hiSel * 8];
    a1.u[0] = *(const u32x4*)&sF[(ar0 + 16) * SF_STRIDE + hiSel * 8];
    a1.u[1] = *(const u32x4*)&sF[(ar0 + 16) * SF_STRIDE + 16 + hiSel * 8];

    // 8 column tiles; each B-frag feeds two WMMAs (rows +0 / +16)
    #pragma unroll
    for (int t = 0; t < 8; ++t) {
      Frag bb;
      const int brow = waveC * 128 + t * 16 + laneLo;   // SimHash row = output col
      bb.u[0] = *(const u32x4*)&sA[brow * SA_STRIDE + hiSel * 8];
      bb.u[1] = *(const u32x4*)&sA[brow * SA_STRIDE + 16 + hiSel * 8];
      acc[0][t] = __builtin_amdgcn_wmma_f32_16x16x32_bf16(
          false, a0.v, false, bb.v, (short)0, acc[0][t], false, false);
      acc[1][t] = __builtin_amdgcn_wmma_f32_16x16x32_bf16(
          false, a1.v, false, bb.v, (short)0, acc[1][t], false, false);
    }
  }

  // epilogue: C/D layout -> lane holds (M = v + 8*hiSel, N = laneLo)
  float* Pout = P + (size_t)blockIdx.y * B_ROWS * K_BITS;
  #pragma unroll
  for (int ar = 0; ar < 2; ++ar) {
    #pragma unroll
    for (int t = 0; t < 8; ++t) {
      #pragma unroll
      for (int v = 0; v < 8; ++v) {
        const int row = rowBlock + waveR * 32 + ar * 16 + hiSel * 8 + v;
        const int col = waveC * 128 + t * 16 + laneLo;
        Pout[(size_t)row * K_BITS + col] = acc[ar][t][v];
      }
    }
  }
}

// ============================================================================
// Kernel 2: reduce the 18 split-K partials and pack sign codes:
//           mask[row][0..7] = pos ballots, mask[row][8..15] = neg ballots
// One wave per row; 8 rows per 256-thread block.
// ============================================================================
__global__ __launch_bounds__(256)
void simhash_reduce_signpack(const float* __restrict__ P,
                             unsigned int* __restrict__ mask) {
  const int wave = threadIdx.x >> 5;
  const int lane = threadIdx.x & 31;
  const int row  = blockIdx.x * 8 + wave;
  #pragma unroll
  for (int w = 0; w < 8; ++w) {
    float v = 0.0f;
    #pragma unroll
    for (int s = 0; s < KSLICES; ++s)
      v += P[(size_t)s * B_ROWS * K_BITS + (size_t)row * K_BITS + w * 32 + lane];
    const unsigned int p = __builtin_amdgcn_ballot_w32(v > 0.0f);
    const unsigned int n = __builtin_amdgcn_ballot_w32(v < 0.0f);
    if (lane == 0) {
      mask[(size_t)row * 16 + w]     = p;
      mask[(size_t)row * 16 + 8 + w] = n;
    }
  }
}

// ============================================================================
// Kernel 3: counts[b] = #{ c : code_b == code_c and no zero signs };
//           out[b] = done[b] / sqrt(counts[b])
// dot(signs_b,signs_c)==256 <=> all 16 mask words equal && (pos|neg)==all-ones.
// Chunk of 256 rows (16 KB) staged in LDS via gfx1250 async global->LDS copy.
// ============================================================================
__global__ __launch_bounds__(256)
void simhash_count_reward(const unsigned int* __restrict__ mask,
                          const float* __restrict__ done,
                          float* __restrict__ out) {
  __shared__ unsigned int sM[256 * 16];    // 16 KB
  const int tid = threadIdx.x;
  const int b   = blockIdx.x * 256 + tid;

  unsigned int m[16];
  unsigned int full = 0xFFFFFFFFu;
  #pragma unroll
  for (int i = 0; i < 16; ++i) m[i] = mask[(size_t)b * 16 + i];
  #pragma unroll
  for (int i = 0; i < 8; ++i) full &= (m[i] | m[8 + i]);
  const int selfOk = (full == 0xFFFFFFFFu) ? 1 : 0;

  int count = 0;
  #pragma unroll 1
  for (int cb = 0; cb < B_ROWS; cb += 256) {
    __syncthreads();                       // previous chunk consumed
    #pragma unroll
    for (int i = 0; i < 4; ++i)
      async_copy_b128(&mask[(size_t)(cb + tid) * 16 + i * 4], &sM[tid * 16 + i * 4]);
    async_wait0();
    __syncthreads();                       // chunk staged by all waves

    #pragma unroll 1
    for (int c = 0; c < 256; ++c) {
      int eq = selfOk;
      #pragma unroll
      for (int i = 0; i < 16; ++i) eq &= (m[i] == sM[c * 16 + i]);
      count += eq;
    }
  }
  out[b] = done[b] * (1.0f / sqrtf((float)count));
}

// ============================================================================
// launch
// ============================================================================
extern "C" void kernel_launch(void* const* d_in, const int* in_sizes, int n_in,
                              void* d_out, int out_size, void* d_ws, size_t ws_size,
                              hipStream_t stream) {
  const float* feats = (const float*)d_in[0];  // [2048, 4*84*84] f32
  const float* done  = (const float*)d_in[1];  // [2048, 1] f32
  const float* Amat  = (const float*)d_in[2];  // [256, 28224] f32
  float* out = (float*)d_out;                  // [2048, 1] f32

  // workspace: split-K partials (18 x 2 MB = 36 MB) + merged sign masks (128 KB)
  char* ws = (char*)d_ws;
  float* P = (float*)ws;
  unsigned int* mask =
      (unsigned int*)(ws + (size_t)KSLICES * B_ROWS * K_BITS * sizeof(float));

  simhash_gemm_bf16_wmma<<<dim3(B_ROWS / BLK_ROWS, KSLICES), dim3(256), 0, stream>>>(feats, Amat, P);
  simhash_reduce_signpack<<<dim3(B_ROWS / 8), dim3(256), 0, stream>>>(P, mask);
  simhash_count_reward<<<dim3(B_ROWS / 256), dim3(256), 0, stream>>>(mask, done, out);
}
